// FeatureGCNProcessor_22582938043089
// MI455X (gfx1250) — compile-verified, matrix operand order checked by reading, not run on hardware
//
#include <hip/hip_runtime.h>
#include <hip/hip_bf16.h>
#include <stdint.h>

// ---------------------------------------------------------------------------
// FeatureGCNProcessor for MI455X (gfx1250, wave32, WMMA)
//   1) transpose BCHW -> node-major features (LDS tiled)
//   2) L2-normalize rows -> bf16 xn; init deg=1 (self loop)
//   3) cosine-sim kNN via v_wmma_f32_16x16x32_bf16 + per-row top-4 (LDS);
//      A fragments register-resident; B tiles double-buffered with
//      global_load_async_to_lds_b128 overlapping the WMMA + scan phase
//   4) dinv = rsqrt(deg)
//   5) f32 GEMM (v_wmma_f32_16x16x4_f32) for x@W1 / h@W2, + global prefetch
//   6) GCN aggregation: self term + edge scatter (native f32 atomics)
//   7) bias+relu (+ final transpose back to BCHW)
// ---------------------------------------------------------------------------

typedef __attribute__((ext_vector_type(2)))  float  v2f;
typedef __attribute__((ext_vector_type(8)))  float  v8f;
typedef __attribute__((ext_vector_type(16))) __bf16 v16bf;
typedef int vsi4 __attribute__((__vector_size__(16)));   // matches builtin param

union BF16Vec { v16bf v; unsigned int u[8]; };

static constexpr int Bimg = 4;
static constexpr int Cdim = 256;
static constexpr int Nimg = 64 * 64;       // 4096 nodes per image
static constexpr int NT   = Bimg * Nimg;   // 16384 nodes total
static constexpr int CH   = 512;
static constexpr int KNN  = 4;

#if defined(__has_builtin)
#  if __has_builtin(__builtin_amdgcn_wmma_f32_16x16x4_f32)
#    define HAVE_WMMA_F32X4 1
#  endif
#  if __has_builtin(__builtin_amdgcn_global_load_async_to_lds_b128) && \
      __has_builtin(__builtin_amdgcn_s_wait_asynccnt)
#    define HAVE_ASYNC_LDS 1
#  endif
#endif
#ifndef HAVE_WMMA_F32X4
#  define HAVE_WMMA_F32X4 0
#endif
#ifndef HAVE_ASYNC_LDS
#  define HAVE_ASYNC_LDS 0
#endif

static __device__ inline unsigned short f2bf(float f) {
  unsigned int u = __float_as_uint(f);
  unsigned int r = u + 0x7FFFu + ((u >> 16) & 1u);   // round-to-nearest-even
  return (unsigned short)(r >> 16);
}

static __device__ inline void top4_insert(float v, int idx, float tv[4], int ti[4]) {
  if (v <= tv[3]) return;                 // fast reject (common case)
  if (v > tv[1]) {
    tv[3] = tv[2]; ti[3] = ti[2];
    tv[2] = tv[1]; ti[2] = ti[1];
    if (v > tv[0]) { tv[1] = tv[0]; ti[1] = ti[0]; tv[0] = v; ti[0] = idx; }
    else           { tv[1] = v;     ti[1] = idx; }
  } else {
    if (v > tv[2]) { tv[3] = tv[2]; ti[3] = ti[2]; tv[2] = v; ti[2] = idx; }
    else           { tv[3] = v;     ti[3] = idx; }
  }
}

// --------------------------- 1) BCHW -> node-major -------------------------
__global__ __launch_bounds__(256) void k_transpose(const float* __restrict__ in,
                                                   float* __restrict__ xout) {
  __shared__ float tile[32][65];
  const int hw0 = blockIdx.x * 64;
  const int c0  = blockIdx.y * 32;
  const int b   = blockIdx.z;
  const int t   = threadIdx.x;
#pragma unroll
  for (int i = 0; i < 8; ++i) {
    int idx = t + i * 256;                // 32 c-rows x 64 hw-cols
    int row = idx >> 6, col = idx & 63;
    tile[row][col] = in[((size_t)(b * Cdim + c0 + row)) * Nimg + hw0 + col];
  }
  __syncthreads();
#pragma unroll
  for (int i = 0; i < 8; ++i) {
    int idx = t + i * 256;                // 64 node-rows x 32 c-cols
    int nrow = idx >> 5, ccol = idx & 31;
    xout[((size_t)(b * Nimg + hw0 + nrow)) * Cdim + c0 + ccol] = tile[ccol][nrow];
  }
}

// --------------------- 2) row L2-normalize -> bf16, deg=1 ------------------
__global__ __launch_bounds__(256) void k_normalize(const float* __restrict__ x,
                                                   unsigned short* __restrict__ xn,
                                                   float* __restrict__ deg) {
  __shared__ float red[256];
  const int n = blockIdx.x, t = threadIdx.x;
  float v = x[(size_t)n * Cdim + t];
  red[t] = v * v;
  __syncthreads();
#pragma unroll
  for (int s = 128; s > 0; s >>= 1) {
    if (t < s) red[t] += red[t + s];
    __syncthreads();
  }
  float inv = 1.0f / fmaxf(sqrtf(red[0]), 1e-12f);
  xn[(size_t)n * Cdim + t] = f2bf(v * inv);
  if (t == 0) deg[n] = 1.0f;              // self loop
}

// ------------- 3) similarity (bf16 WMMA) + per-row top-4 neighbors ---------
// block: 256 threads (8 waves). Block owns 128 rows of one image (16/wave).
// A fragments (16 rows x 256 K per wave) are hoisted into registers once.
// B tiles are double-buffered: each iteration waits for the async batch
// issued last iteration, then issues the next tile's async global->LDS copy
// before the WMMA chain, so the DMA overlaps compute. 8x
// v_wmma_f32_16x16x32_bf16 per tile; sim tile spilled to LDS and scanned:
// each lane maintains a top-4 list for one row / 8 columns.
__global__ __launch_bounds__(256) void k_simtopk(const unsigned short* __restrict__ xn,
                                                 int* __restrict__ tgt,
                                                 float* __restrict__ deg) {
  extern __shared__ char smem[];
  unsigned short* ldsA  = (unsigned short*)smem;            // 128 rows * 272
  unsigned short* ldsB0 = ldsA + 128 * 272;                 // 16 rows * 272
  unsigned short* ldsB1 = ldsB0 + 16 * 272;                 // 16 rows * 272
  float*          ldsS  = (float*)(ldsB1 + 16 * 272);       // 8 waves * 16*17
  float*          ldsMV = ldsS + 8 * 16 * 17;               // 8 * 128
  int*            ldsMI = (int*)(ldsMV + 8 * 128);          // 8 * 128

  const int t    = threadIdx.x;
  const int wv   = t >> 5;
  const int l    = t & 31;
  const int m    = l & 15;
  const int half = l >> 4;
  const int img  = blockIdx.y;
  const int r0   = blockIdx.x * 128;
  const uint4* xnq = (const uint4*)xn;    // 32 uint4 per 256-bf16 row

  // per-thread B staging coordinates (2 x uint4 per thread covers 16x256 bf16)
  const int brow0 = t >> 5,        bq0 = t & 31;
  const int brow1 = (t + 256) >> 5, bq1 = t & 31;

  // stage A: 128 rows x 256 bf16 (4096 uint4)
#pragma unroll
  for (int j = 0; j < 16; ++j) {
    int idx = t + j * 256;
    int row = idx >> 5, q = idx & 31;
    uint4 v = xnq[((size_t)(img * Nimg + r0 + row)) * 32 + q];
    *(uint4*)(ldsA + row * 272 + q * 8) = v;
  }
  __syncthreads();

  // hoist this wave's A fragments into registers (loop-invariant)
  const unsigned short* arow = ldsA + (wv * 16 + m) * 272;
  BF16Vec afrag[8];
#pragma unroll
  for (int s = 0; s < 8; ++s) {
    const int k0 = 32 * s + 8 * half;     // element e -> K = (e&7)+8*half+16*(e>>3)+32*s
    uint4 alo = *(const uint4*)(arow + k0);
    uint4 ahi = *(const uint4*)(arow + k0 + 16);
    afrag[s].u[0] = alo.x; afrag[s].u[1] = alo.y;
    afrag[s].u[2] = alo.z; afrag[s].u[3] = alo.w;
    afrag[s].u[4] = ahi.x; afrag[s].u[5] = ahi.y;
    afrag[s].u[6] = ahi.z; afrag[s].u[7] = ahi.w;
  }

  // issue B stage for a tile into a buffer (async when available)
  auto issueB = [&](int tile, unsigned short* buf) {
    const int col0 = tile * 16;
    const uint4* s0 = &xnq[((size_t)(img * Nimg + col0 + brow0)) * 32 + bq0];
    const uint4* s1 = &xnq[((size_t)(img * Nimg + col0 + brow1)) * 32 + bq1];
    unsigned short* d0 = buf + brow0 * 272 + bq0 * 8;
    unsigned short* d1 = buf + brow1 * 272 + bq1 * 8;
#if HAVE_ASYNC_LDS
    __builtin_amdgcn_global_load_async_to_lds_b128((vsi4*)s0, (vsi4*)d0, 0, 0);
    __builtin_amdgcn_global_load_async_to_lds_b128((vsi4*)s1, (vsi4*)d1, 0, 0);
#else
    *(uint4*)d0 = *s0;
    *(uint4*)d1 = *s1;
#endif
  };

  float tv[4] = {-1e30f, -1e30f, -1e30f, -1e30f};
  int   ti[4] = {0, 0, 0, 0};
  const int myrow = r0 + wv * 16 + m;     // row this lane scans (within image)
  float* simw = ldsS + wv * (16 * 17);
  constexpr int NTILE = Nimg / 16;

  issueB(0, ldsB0);                        // prologue: fill buffer 0

  for (int tile = 0; tile < NTILE; ++tile) {
#if HAVE_ASYNC_LDS
    __builtin_amdgcn_s_wait_asynccnt(0);   // batch issued last iteration done
#endif
    __syncthreads();                       // ... in every wave
    if (tile + 1 < NTILE)                  // overlap next DMA with compute
      issueB(tile + 1, (tile & 1) ? ldsB0 : ldsB1);

    const unsigned short* browC = ((tile & 1) ? ldsB1 : ldsB0) + m * 272;
    // bulk-load all B fragments so ds_loads pipeline behind one wait
    uint4 bfr[16];
#pragma unroll
    for (int s = 0; s < 8; ++s) {
      const int k0 = 32 * s + 8 * half;
      bfr[2 * s]     = *(const uint4*)(browC + k0);
      bfr[2 * s + 1] = *(const uint4*)(browC + k0 + 16);
    }
    v8f acc = {0.f, 0.f, 0.f, 0.f, 0.f, 0.f, 0.f, 0.f};
#pragma unroll
    for (int s = 0; s < 8; ++s) {
      BF16Vec Bv;
      Bv.u[0] = bfr[2 * s].x;     Bv.u[1] = bfr[2 * s].y;
      Bv.u[2] = bfr[2 * s].z;     Bv.u[3] = bfr[2 * s].w;
      Bv.u[4] = bfr[2 * s + 1].x; Bv.u[5] = bfr[2 * s + 1].y;
      Bv.u[6] = bfr[2 * s + 1].z; Bv.u[7] = bfr[2 * s + 1].w;
      acc = __builtin_amdgcn_wmma_f32_16x16x32_bf16(false, afrag[s].v, false, Bv.v,
                                                    (short)0, acc, false, false);
    }
    // spill D tile: lane holds rows r+8*half, col m
#pragma unroll
    for (int r = 0; r < 8; ++r) simw[(r + 8 * half) * 17 + m] = acc[r];
    __syncthreads();
    // scan: lane owns row m, cols 8*half..8*half+7
    const int col0 = tile * 16;
#pragma unroll
    for (int j = 0; j < 8; ++j) {
      int c = 8 * half + j;
      int gcol = col0 + c;
      float v = simw[m * 17 + c];
      if (gcol != myrow) top4_insert(v, gcol, tv, ti);
    }
  }

  // merge the two half-wave partial lists per row
  __syncthreads();
  {
    int base = wv * 128 + m * 8 + half * 4;
#pragma unroll
    for (int j = 0; j < 4; ++j) { ldsMV[base + j] = tv[j]; ldsMI[base + j] = ti[j]; }
  }
  __syncthreads();
  if (half == 0) {
    int pb = wv * 128 + m * 8 + 4;
#pragma unroll
    for (int j = 0; j < 4; ++j) top4_insert(ldsMV[pb + j], ldsMI[pb + j], tv, ti);
    int node = img * Nimg + myrow;
#pragma unroll
    for (int j = 0; j < 4; ++j) {
      tgt[node * KNN + j] = img * Nimg + ti[j];
      unsafeAtomicAdd(&deg[img * Nimg + ti[j]], 1.0f);
    }
  }
}

// ------------------------------ 4) dinv ------------------------------------
__global__ void k_dinv(const float* __restrict__ deg, float* __restrict__ dinv) {
  int i = blockIdx.x * 256 + threadIdx.x;
  if (i < NT) { float d = deg[i]; dinv[i] = (d > 0.f) ? rsqrtf(d) : 0.f; }
}

// ----------------- 5) f32 GEMM: C[MxN] = A[MxK] * W[KxN] -------------------
// block = 256 threads (8 waves); each block computes a 16x128 tile; each wave
// one 16x16 tile via K/4 V_WMMA_F32_16X16X4_F32 steps from LDS.
__global__ __launch_bounds__(256) void k_gemm(const float* __restrict__ A,
                                              const float* __restrict__ W,
                                              float* __restrict__ C,
                                              int M, int K, int N) {
  __shared__ float ldsA[16 * 68];     // 16 x 64, stride 68
  __shared__ float ldsB[64 * 136];    // 64 x 128, stride 136
  const int t = threadIdx.x, wv = t >> 5, l = t & 31, m = l & 15, half = l >> 4;
  const int m0 = blockIdx.x * 16;
  const int n0 = blockIdx.y * 128;
  v8f acc = {0.f, 0.f, 0.f, 0.f, 0.f, 0.f, 0.f, 0.f};

  const int arow = t >> 4, acol4 = (t & 15) * 4;

  for (int k0 = 0; k0 < K; k0 += 64) {
    { // A: 16 rows x 64 cols, one float4 per thread
      float4 av = *(const float4*)&A[(size_t)(m0 + arow) * K + k0 + acol4];
      *(float4*)&ldsA[arow * 68 + acol4] = av;
    }
#pragma unroll
    for (int j = 0; j < 8; ++j) { // B: 64 rows x 128 cols
      int idx = t + j * 256;
      int row = idx >> 5, col4 = (idx & 31) * 4;
      float4 bv = *(const float4*)&W[(size_t)(k0 + row) * N + n0 + col4];
      *(float4*)&ldsB[row * 136 + col4] = bv;
    }
    if (k0 + 64 < K) {  // speculative prefetch of next K-chunk (global_prefetch_b8)
      __builtin_prefetch(&A[(size_t)(m0 + arow) * K + k0 + 64 + acol4], 0, 3);
      __builtin_prefetch(&W[(size_t)(k0 + 64 + (t >> 2)) * N + n0 + (t & 3) * 32], 0, 3);
    }
    __syncthreads();
#if HAVE_WMMA_F32X4
#pragma unroll
    for (int s = 0; s < 16; ++s) {
      int kk = 4 * s + 2 * half;          // element r -> K = kk + r
      v2f a, b;
      a.x = ldsA[m * 68 + kk];
      a.y = ldsA[m * 68 + kk + 1];
      b.x = ldsB[kk * 136 + wv * 16 + m];
      b.y = ldsB[(kk + 1) * 136 + wv * 16 + m];
      acc = __builtin_amdgcn_wmma_f32_16x16x4_f32(false, a, false, b,
                                                  (short)0, acc, false, false);
    }
#else
    for (int kk = 0; kk < 64; ++kk) {
      float bcol = ldsB[kk * 136 + wv * 16 + m];
#pragma unroll
      for (int r = 0; r < 8; ++r)
        acc[r] += ldsA[(r + 8 * half) * 68 + kk] * bcol;
    }
#endif
    __syncthreads();
  }
#pragma unroll
  for (int r = 0; r < 8; ++r)
    C[(size_t)(m0 + r + 8 * half) * N + n0 + wv * 16 + m] = acc[r];
}

// -------------------- 6) GCN aggregation (self + edges) --------------------
__global__ void k_aggself(const float* __restrict__ xw, const float* __restrict__ dinv,
                          float* __restrict__ outb, int F) {
  int n = blockIdx.x;
  int c = blockIdx.y * 256 + threadIdx.x;
  float w = dinv[n]; w = w * w;
  outb[(size_t)n * F + c] = xw[(size_t)n * F + c] * w;
}

__global__ void k_aggedge(const float* __restrict__ xw, const int* __restrict__ tgt,
                          const float* __restrict__ dinv, float* __restrict__ outb, int F) {
  int e = blockIdx.x;
  int s = e >> 2;                          // src node (global id)
  int tg = tgt[e];                         // dst node (global id)
  float w = dinv[s] * dinv[tg];
  for (int c = threadIdx.x; c < F; c += 256)
    unsafeAtomicAdd(&outb[(size_t)tg * F + c], xw[(size_t)s * F + c] * w);
}

__global__ void k_biasrelu(float* __restrict__ h, const float* __restrict__ bias, int Fmask) {
  size_t i = (size_t)blockIdx.x * 256 + threadIdx.x;
  int c = (int)(i & (size_t)Fmask);
  h[i] = fmaxf(h[i] + bias[c], 0.f);
}

// ------------- 7) bias+relu + node-major -> BCHW transpose -----------------
__global__ __launch_bounds__(256) void k_final(const float* __restrict__ agg,
                                               const float* __restrict__ b2,
                                               float* __restrict__ out) {
  __shared__ float tile[64][33];
  const int hw0 = blockIdx.x * 64;
  const int c0  = blockIdx.y * 32;
  const int b   = blockIdx.z;
  const int t   = threadIdx.x;
#pragma unroll
  for (int i = 0; i < 8; ++i) {
    int idx = t + i * 256;                // 64 node-rows x 32 c-cols
    int nrow = idx >> 5, ccol = idx & 31;
    float v = agg[((size_t)(b * Nimg + hw0 + nrow)) * Cdim + c0 + ccol] + b2[c0 + ccol];
    tile[nrow][ccol] = fmaxf(v, 0.f);
  }
  __syncthreads();
#pragma unroll
  for (int i = 0; i < 8; ++i) {
    int idx = t + i * 256;                // 32 c-rows x 64 hw-cols
    int crow = idx >> 6, hcol = idx & 63;
    out[((size_t)(b * Cdim + c0 + crow)) * Nimg + hw0 + hcol] = tile[hcol][crow];
  }
}

// ---------------------------------------------------------------------------
extern "C" void kernel_launch(void* const* d_in, const int* in_sizes, int n_in,
                              void* d_out, int out_size, void* d_ws, size_t ws_size,
                              hipStream_t stream) {
  const float* fmaps = (const float*)d_in[0];
  const float* W1    = (const float*)d_in[1];
  const float* b1    = (const float*)d_in[2];
  const float* W2    = (const float*)d_in[3];
  const float* b2    = (const float*)d_in[4];
  float* out = (float*)d_out;

  char* ws = (char*)d_ws;
  float*          x    = (float*)ws;          ws += (size_t)NT * Cdim * 4;
  unsigned short* xn   = (unsigned short*)ws; ws += (size_t)NT * Cdim * 2;
  int*            tgt  = (int*)ws;            ws += (size_t)NT * KNN * 4;
  float*          deg  = (float*)ws;          ws += (size_t)NT * 4;
  float*          dinv = (float*)ws;          ws += (size_t)NT * 4;
  float*          xw1  = (float*)ws;          ws += (size_t)NT * CH * 4;
  float*          h1   = (float*)ws;          ws += (size_t)NT * CH * 4;
  float*          xw2  = (float*)ws;          ws += (size_t)NT * Cdim * 4;
  float*          h2   = (float*)ws;          ws += (size_t)NT * Cdim * 4;

  k_transpose<<<dim3(Nimg / 64, Cdim / 32, Bimg), 256, 0, stream>>>(fmaps, x);
  k_normalize<<<NT, 256, 0, stream>>>(x, xn, deg);

  const size_t simsh = (size_t)128 * 272 * 2 + 2 * 16 * 272 * 2
                     + 8 * 16 * 17 * 4 + 8 * 128 * 4 + 8 * 128 * 4;
  k_simtopk<<<dim3(Nimg / 128, Bimg), 256, simsh, stream>>>(xn, tgt, deg);
  k_dinv<<<NT / 256, 256, 0, stream>>>(deg, dinv);

  // layer 1: h1 = relu(agg(x @ W1) + b1)
  k_gemm<<<dim3(NT / 16, CH / 128), 256, 0, stream>>>(x, W1, xw1, NT, Cdim, CH);
  k_aggself<<<dim3(NT, CH / 256), 256, 0, stream>>>(xw1, dinv, h1, CH);
  k_aggedge<<<NT * KNN, 256, 0, stream>>>(xw1, tgt, dinv, h1, CH);
  k_biasrelu<<<(NT * CH) / 256, 256, 0, stream>>>(h1, b1, CH - 1);

  // layer 2: out = relu(agg(h1 @ W2) + b2), back to BCHW
  k_gemm<<<dim3(NT / 16, Cdim / 128), 256, 0, stream>>>(h1, W2, xw2, NT, CH, Cdim);
  k_aggself<<<dim3(NT, Cdim / 256), 256, 0, stream>>>(xw2, dinv, h2, Cdim);
  k_aggedge<<<NT * KNN, 256, 0, stream>>>(xw2, tgt, dinv, h2, Cdim);
  k_final<<<dim3(Nimg / 64, Cdim / 32, Bimg), 256, 0, stream>>>(h2, b2, out);
}
